// CosineCrossAttention_6786048328368
// MI455X (gfx1250) — matrix operand order repaired and checked by reading, not compiled, not run on hardware
//
#include <hip/hip_runtime.h>
#include <hip/hip_bf16.h>

// ---------------------------------------------------------------------------
// CosineCrossAttention on MI455X (gfx1250, wave32, WMMA bf16 path)
//   B=4, L=2048, D=1024, H=16, Dh=64, tau=0.07
// Pipeline:
//   1) cvt_f32_bf16   one-shot fp32->bf16 for activations + weights
//   2) proj_gemm<2>   WMMA GEMM + bias + per-head L2 norm -> bf16 [B,H,L,64]
//      proj_gemm<3>   WMMA GEMM + bias, TRANSPOSED out    -> bf16 [B,H,64,L]
//   3) flash_attn     online-softmax attention; K and V^T tiles staged with
//                     double-buffered global_load_async_to_lds_b128
//   4) proj_gemm<0>   WMMA GEMM + bias -> fp32 d_out
// ---------------------------------------------------------------------------

#define B_  4
#define L_  2048
#define D_  1024
#define H_  16
#define DH_ 64
#define M_  (B_ * L_)          // 8192 rows through the projections

typedef __attribute__((ext_vector_type(16))) __bf16 v16bf;
typedef __attribute__((ext_vector_type(8)))  __bf16 bf16x8;
typedef __attribute__((ext_vector_type(8)))  float  v8f;
typedef __attribute__((ext_vector_type(4)))  float  f32x4;

static __device__ __forceinline__ __bf16 f2bf(float f) {
  return (__bf16)f;   // native fptrunc (RTNE) on gfx1250
}

static __device__ __forceinline__ v8f wmma_bf16(v16bf a, v16bf b, v8f c) {
  // D = A(16x32) * B(32x16) + C, fp32 accumulate
  return __builtin_amdgcn_wmma_f32_16x16x32_bf16(
      false, a, false, b, (short)0, c, false, false);
}

// Async global -> LDS copy of 16 bytes per lane (ASYNCcnt-tracked).
static __device__ __forceinline__ void async_copy_b128(void* lds_dst,
                                                       const void* gsrc) {
  // generic pointer to __shared__ data: low 32 bits == LDS byte offset
  unsigned loff = (unsigned)(size_t)lds_dst;
  asm volatile("global_load_async_to_lds_b128 %0, %1, off"
               :: "v"(loff), "v"(gsrc)
               : "memory");
}
static __device__ __forceinline__ void wait_async0() {
  asm volatile("s_wait_asynccnt 0x0" ::: "memory");
}
static __device__ __forceinline__ void wait_async1() {
  asm volatile("s_wait_asynccnt 0x1" ::: "memory");
}
static __device__ __forceinline__ void wait_async4() {
  asm volatile("s_wait_asynccnt 0x4" ::: "memory");
}

// ---------------------------------------------------------------------------
// Elementwise fp32 -> bf16 (8 elements / thread). n must be a multiple of 8.
// ---------------------------------------------------------------------------
__global__ __launch_bounds__(256)
void cvt_f32_bf16(const float* __restrict__ in, unsigned short* __restrict__ outU,
                  int n) {
  __bf16* out = reinterpret_cast<__bf16*>(outU);
  const int i = (blockIdx.x * 256 + threadIdx.x) * 8;
  if (i + 8 > n) return;
  f32x4 a = *(const f32x4*)(in + i);
  f32x4 b = *(const f32x4*)(in + i + 4);
  bf16x8 o;
#pragma unroll
  for (int j = 0; j < 4; ++j) { o[j] = f2bf(a[j]); o[4 + j] = f2bf(b[j]); }
  *(bf16x8*)(out + i) = o;
}

// ---------------------------------------------------------------------------
// GEMM: acc[M,N] = A[M,K] @ W[N,K]^T + bias[N], bf16 operands, fp32 acc.
// Block: 256 threads = 8 waves; wave w owns rows [(by*8+w)*16, +16);
// block owns cols [bx*64, +64) == exactly one head when N == 1024.
// W k-slab (64 cols x 32 k) double-buffered in LDS via async-to-LDS.
// MODE 0: fp32 out [M,N]                                (output projection)
// MODE 2: bf16 out [B,H,L,64], per-(row,head) L2 norm   (Q/K projections)
// MODE 3: bf16 out [B,H,64,L]  (transposed via LDS)     (V projection)
// ---------------------------------------------------------------------------
template <int MODE>
__global__ __launch_bounds__(256)
void proj_gemm(const unsigned short* __restrict__ Au,
               const unsigned short* __restrict__ Wu,
               const float* __restrict__ bias,
               float* __restrict__ outF,
               unsigned short* __restrict__ outBu,
               int M, int N, int K) {
  const __bf16* A = reinterpret_cast<const __bf16*>(Au);
  const __bf16* W = reinterpret_cast<const __bf16*>(Wu);
  __bf16* outB = reinterpret_cast<__bf16*>(outBu);

  __shared__ __bf16 Wt[2][64][40];   // double-buffered; 80 B rows (16B align)

  const int lane = threadIdx.x & 31;
  const int wave = threadIdx.x >> 5;
  const int lo = lane & 15;
  const int hi = lane >> 4;
  const int mrow0 = (blockIdx.y * 8 + wave) * 16;
  const int ncol0 = blockIdx.x * 64;

  v8f acc[4];
#pragma unroll
  for (int t = 0; t < 4; ++t)
#pragma unroll
    for (int r = 0; r < 8; ++r) acc[t][r] = 0.0f;

  // cooperative W staging: 64 rows x 32 k per slab, 8 bf16 (16 B) per thread
  const int wrow = threadIdx.x >> 2;        // 0..63
  const int kch  = (threadIdx.x & 3) * 8;   // 0,8,16,24
  const __bf16* wbase = W + (size_t)(ncol0 + wrow) * K + kch;

  const int NK = K >> 5;
  async_copy_b128(&Wt[0][wrow][kch], wbase);          // prologue: slab 0

  for (int i = 0; i < NK; ++i) {
    const int cur = i & 1;
    if (i + 1 < NK) {
      async_copy_b128(&Wt[cur ^ 1][wrow][kch], wbase + (i + 1) * 32);
      wait_async1();          // slab i resident; slab i+1 may be in flight
    } else {
      wait_async0();
    }
    __syncthreads();

    const int kb = i * 32;
    // A fragment: row (mrow0+lo); K chunks {kb+8*hi+j, kb+16+8*hi+j}
    v16bf a;
    {
      const __bf16* ap = A + (size_t)(mrow0 + lo) * K + kb + 8 * hi;
      bf16x8 c0 = *(const bf16x8*)(ap);
      bf16x8 c1 = *(const bf16x8*)(ap + 16);
#pragma unroll
      for (int j = 0; j < 8; ++j) { a[j] = c0[j]; a[8 + j] = c1[j]; }
    }
#pragma unroll
    for (int t = 0; t < 4; ++t) {
      const __bf16* wp = &Wt[cur][16 * t + lo][8 * hi];
      bf16x8 c0 = *(const bf16x8*)(wp);
      bf16x8 c1 = *(const bf16x8*)(wp + 16);
      v16bf b;
#pragma unroll
      for (int j = 0; j < 8; ++j) { b[j] = c0[j]; b[8 + j] = c1[j]; }
      acc[t] = wmma_bf16(a, b, acc[t]);
    }
    __syncthreads();          // release slab buffer before next prefetch
  }

  // epilogue: bias (+ optional per-row L2 norm over this head's 64 cols)
  float val[4][8];
#pragma unroll
  for (int t = 0; t < 4; ++t) {
    const float bv = bias[ncol0 + 16 * t + lo];
#pragma unroll
    for (int r = 0; r < 8; ++r) val[t][r] = acc[t][r] + bv;
  }

  if constexpr (MODE == 2) {
#pragma unroll
    for (int r = 0; r < 8; ++r) {
      float ss = 0.0f;
#pragma unroll
      for (int t = 0; t < 4; ++t) ss += val[t][r] * val[t][r];
#pragma unroll
      for (int mask = 8; mask >= 1; mask >>= 1) ss += __shfl_xor(ss, mask, 16);
      const float inv = 1.0f / fmaxf(sqrtf(ss), 1e-12f);
#pragma unroll
      for (int t = 0; t < 4; ++t) val[t][r] *= inv;
    }
  }

  if constexpr (MODE == 0) {
#pragma unroll
    for (int t = 0; t < 4; ++t) {
      const int col = ncol0 + 16 * t + lo;
#pragma unroll
      for (int r = 0; r < 8; ++r) {
        const int row = mrow0 + r + 8 * hi;
        outF[(size_t)row * N + col] = val[t][r];
      }
    }
  } else if constexpr (MODE == 2) {
    const int h = blockIdx.x;   // one head per block column (N == 1024)
#pragma unroll
    for (int t = 0; t < 4; ++t) {
      const int col = 16 * t + lo;
#pragma unroll
      for (int r = 0; r < 8; ++r) {
        const int m = mrow0 + r + 8 * hi;      // row in [0, B*L)
        const int b = m >> 11;                 // / L_
        const int l = m & (L_ - 1);
        outB[(((size_t)b * H_ + h) * L_ + l) * DH_ + col] = f2bf(val[t][r]);
      }
    }
  } else {  // MODE == 3: transpose through LDS, coalesced b128 stores
    __shared__ __bf16 Tb[128][72];
    const int rowl0 = wave * 16;
#pragma unroll
    for (int t = 0; t < 4; ++t)
#pragma unroll
      for (int r = 0; r < 8; ++r)
        Tb[rowl0 + r + 8 * hi][16 * t + lo] = f2bf(val[t][r]);
    __syncthreads();

    const int h   = blockIdx.x;
    const int d   = threadIdx.x >> 2;          // 0..63
    const int seg = (threadIdx.x & 3) * 32;    // token segment
    const int m0  = blockIdx.y * 128;          // block's first row (token)
    const int b   = m0 >> 11;
    const int l0  = m0 & (L_ - 1);
    __bf16* dst = outB + (((size_t)b * H_ + h) * DH_ + d) * L_ + l0 + seg;
#pragma unroll
    for (int c = 0; c < 4; ++c) {
      bf16x8 o;
#pragma unroll
      for (int j = 0; j < 8; ++j) o[j] = Tb[seg + c * 8 + j][d];
      *(bf16x8*)(dst + c * 8) = o;
    }
  }
}

// ---------------------------------------------------------------------------
// Flash attention per (b,h). Block = 256 threads (8 waves); 128 query rows
// per block (16/wave); KV blocks of 64 keys. K [L,64] and V^T [64,L] tiles
// are double-buffered in LDS via async-to-LDS b128 (4 issues/thread/tile,
// s_wait_asynccnt 4 overlaps the next tile's DMA with WMMA work).
// P round-trips through LDS for C/D -> A layout. Output bf16 [B, L, H*64].
// ---------------------------------------------------------------------------
__global__ __launch_bounds__(256)
void flash_attn(const unsigned short* __restrict__ Qu,
                const unsigned short* __restrict__ Ku,
                const unsigned short* __restrict__ Vtu,
                unsigned short* __restrict__ Ou) {
  constexpr float invTau = 1.0f / 0.07f;
  __shared__ __bf16 Ks[2][64][80];   // [key][d]
  __shared__ __bf16 Vs[2][64][80];   // [d][key]  (global layout already ^T)
  __shared__ __bf16 Pb[8][16][80];   // per-wave P staging

  const __bf16* Q  = reinterpret_cast<const __bf16*>(Qu);
  const __bf16* K  = reinterpret_cast<const __bf16*>(Ku);
  const __bf16* Vt = reinterpret_cast<const __bf16*>(Vtu);
  __bf16* O = reinterpret_cast<__bf16*>(Ou);

  const int lane = threadIdx.x & 31;
  const int wave = threadIdx.x >> 5;
  const int lo = lane & 15;
  const int hi = lane >> 4;
  const int bh = blockIdx.y;                        // b*H + h
  const int q0 = blockIdx.x * 128 + wave * 16;      // wave's first query row

  const __bf16* Qb  = Q  + (size_t)bh * L_ * DH_;
  const __bf16* Kb  = K  + (size_t)bh * L_ * DH_;
  const __bf16* Vtb = Vt + (size_t)bh * DH_ * L_;

  // staging assignment: row = tid>>2 (key for K, d for V), 16-col chunk
  const int srow = threadIdx.x >> 2;
  const int scol = (threadIdx.x & 3) * 16;
  auto stageKV = [&](int buf, int kb0) {
    const __bf16* kp = Kb + (size_t)(kb0 + srow) * DH_ + scol;
    const __bf16* vp = Vtb + (size_t)srow * L_ + kb0 + scol;
    async_copy_b128(&Ks[buf][srow][scol],     kp);
    async_copy_b128(&Ks[buf][srow][scol + 8], kp + 8);
    async_copy_b128(&Vs[buf][srow][scol],     vp);
    async_copy_b128(&Vs[buf][srow][scol + 8], vp + 8);
  };

  // Q A-fragments for the two 32-wide K-steps of head_dim=64
  v16bf qf[2];
#pragma unroll
  for (int s = 0; s < 2; ++s) {
    const __bf16* qp = Qb + (size_t)(q0 + lo) * DH_ + 32 * s + 8 * hi;
    bf16x8 c0 = *(const bf16x8*)(qp);
    bf16x8 c1 = *(const bf16x8*)(qp + 16);
#pragma unroll
    for (int j = 0; j < 8; ++j) { qf[s][j] = c0[j]; qf[s][8 + j] = c1[j]; }
  }

  float mrow[8], lrow[8];
  v8f oacc[4];
#pragma unroll
  for (int r = 0; r < 8; ++r) { mrow[r] = -3.0e38f; lrow[r] = 0.0f; }
#pragma unroll
  for (int t = 0; t < 4; ++t)
#pragma unroll
    for (int r = 0; r < 8; ++r) oacc[t][r] = 0.0f;

  constexpr int NBLK = L_ / 64;
  stageKV(0, 0);                         // prologue

  for (int i = 0; i < NBLK; ++i) {
    const int cur = i & 1;
    if (i + 1 < NBLK) {
      stageKV(cur ^ 1, (i + 1) * 64);
      wait_async4();   // tile i resident; tile i+1 DMA overlaps compute
    } else {
      wait_async0();
    }
    __syncthreads();

    // ---- S = Q K^T (4 key tiles of 16) ----
    v8f sc[4];
#pragma unroll
    for (int t = 0; t < 4; ++t) {
#pragma unroll
      for (int r = 0; r < 8; ++r) sc[t][r] = 0.0f;
#pragma unroll
      for (int s = 0; s < 2; ++s) {
        const __bf16* kp = &Ks[cur][16 * t + lo][32 * s + 8 * hi];
        bf16x8 c0 = *(const bf16x8*)(kp);
        bf16x8 c1 = *(const bf16x8*)(kp + 16);
        v16bf b;
#pragma unroll
        for (int j = 0; j < 8; ++j) { b[j] = c0[j]; b[8 + j] = c1[j]; }
        sc[t] = wmma_bf16(qf[s], b, sc[t]);
      }
    }

    // ---- online softmax (row = r + 8*hi; half-wave shuffles, width 16) ----
    float scale_r[8];
#pragma unroll
    for (int r = 0; r < 8; ++r) {
      float mx = -3.0e38f;
#pragma unroll
      for (int t = 0; t < 4; ++t) mx = fmaxf(mx, sc[t][r]);
      mx *= invTau;
#pragma unroll
      for (int mask = 8; mask >= 1; mask >>= 1)
        mx = fmaxf(mx, __shfl_xor(mx, mask, 16));
      const float mnew = fmaxf(mrow[r], mx);
      const float rescale = __expf(mrow[r] - mnew);
      float rs = 0.0f;
#pragma unroll
      for (int t = 0; t < 4; ++t) {
        const float p = __expf(sc[t][r] * invTau - mnew);
        sc[t][r] = p;
        rs += p;
      }
#pragma unroll
      for (int mask = 8; mask >= 1; mask >>= 1) rs += __shfl_xor(rs, mask, 16);
      lrow[r] = lrow[r] * rescale + rs;
      mrow[r] = mnew;
      scale_r[r] = rescale;
    }
#pragma unroll
    for (int t = 0; t < 4; ++t)
#pragma unroll
      for (int r = 0; r < 8; ++r) oacc[t][r] *= scale_r[r];

    // ---- stage P (D layout -> LDS), then reload as A fragments ----
#pragma unroll
    for (int t = 0; t < 4; ++t)
#pragma unroll
      for (int r = 0; r < 8; ++r)
        Pb[wave][8 * hi + r][16 * t + lo] = f2bf(sc[t][r]);

#pragma unroll
    for (int s = 0; s < 2; ++s) {
      const __bf16* pp = &Pb[wave][lo][32 * s + 8 * hi];
      bf16x8 c0 = *(const bf16x8*)(pp);
      bf16x8 c1 = *(const bf16x8*)(pp + 16);
      v16bf a;
#pragma unroll
      for (int j = 0; j < 8; ++j) { a[j] = c0[j]; a[8 + j] = c1[j]; }
#pragma unroll
      for (int t = 0; t < 4; ++t) {
        const __bf16* vp = &Vs[cur][16 * t + lo][32 * s + 8 * hi];
        bf16x8 d0 = *(const bf16x8*)(vp);
        bf16x8 d1 = *(const bf16x8*)(vp + 16);
        v16bf b;
#pragma unroll
        for (int j = 0; j < 8; ++j) { b[j] = d0[j]; b[8 + j] = d1[j]; }
        oacc[t] = wmma_bf16(a, b, oacc[t]);
      }
    }
    __syncthreads();          // release KV buffer before next prefetch
  }

  // ---- epilogue: O / l into merged bf16 [B, L, H*64] ----
  const int b = bh >> 4, h = bh & 15;
#pragma unroll
  for (int t = 0; t < 4; ++t) {
    const int col = h * DH_ + 16 * t + lo;
#pragma unroll
    for (int r = 0; r < 8; ++r) {
      const int qrow = q0 + 8 * hi + r;
      O[((size_t)(b * L_ + qrow)) * D_ + col] = f2bf(oacc[t][r] / lrow[r]);
    }
  }
}

// ---------------------------------------------------------------------------
extern "C" void kernel_launch(void* const* d_in, const int* in_sizes, int n_in,
                              void* d_out, int out_size, void* d_ws, size_t ws_size,
                              hipStream_t stream) {
  (void)in_sizes; (void)n_in; (void)out_size; (void)ws_size;

  const float* q  = (const float*)d_in[0];
  const float* k  = (const float*)d_in[1];
  const float* v  = (const float*)d_in[2];
  const float* Wq = (const float*)d_in[3];
  const float* bq = (const float*)d_in[4];
  const float* Wk = (const float*)d_in[5];
  const float* bk = (const float*)d_in[6];
  const float* Wv = (const float*)d_in[7];
  const float* bv = (const float*)d_in[8];
  const float* Wo = (const float*)d_in[9];
  const float* bo = (const float*)d_in[10];
  float* out = (float*)d_out;

  char* ws = (char*)d_ws;
  const size_t ACT_B  = (size_t)M_ * D_ * 2;   // bf16 activations: 16 MiB
  const size_t WGT_B  = (size_t)D_ * D_ * 2;   // bf16 weights:      2 MiB
  unsigned short* qb  = (unsigned short*)(ws);
  unsigned short* kb  = (unsigned short*)(ws + ACT_B);
  unsigned short* vb  = (unsigned short*)(ws + 2 * ACT_B);
  unsigned short* Wqb = (unsigned short*)(ws + 3 * ACT_B);
  unsigned short* Wkb = (unsigned short*)(ws + 3 * ACT_B + WGT_B);
  unsigned short* Wvb = (unsigned short*)(ws + 3 * ACT_B + 2 * WGT_B);
  unsigned short* Wob = (unsigned short*)(ws + 3 * ACT_B + 3 * WGT_B);
  unsigned short* Qh  = (unsigned short*)(ws + 3 * ACT_B + 4 * WGT_B);
  unsigned short* Kh  = (unsigned short*)(ws + 4 * ACT_B + 4 * WGT_B);
  unsigned short* VhT = (unsigned short*)(ws + 5 * ACT_B + 4 * WGT_B);
  unsigned short* Ab  = qb;   // attention output reuses qb after Q projection

  const int nAct = M_ * D_;        // 8388608
  const int nWgt = D_ * D_;        // 1048576
  dim3 cvtBlk(256);
  dim3 cvtActGrid(nAct / (8 * 256));
  dim3 cvtWgtGrid(nWgt / (8 * 256));
  dim3 gemmGrid(D_ / 64, M_ / 128), gemmBlk(256);
  dim3 faGrid(L_ / 128, B_ * H_),   faBlk(256);

  cvt_f32_bf16<<<cvtActGrid, cvtBlk, 0, stream>>>(q,  qb,  nAct);
  cvt_f32_bf16<<<cvtActGrid, cvtBlk, 0, stream>>>(k,  kb,  nAct);
  cvt_f32_bf16<<<cvtActGrid, cvtBlk, 0, stream>>>(v,  vb,  nAct);
  cvt_f32_bf16<<<cvtWgtGrid, cvtBlk, 0, stream>>>(Wq, Wqb, nWgt);
  cvt_f32_bf16<<<cvtWgtGrid, cvtBlk, 0, stream>>>(Wk, Wkb, nWgt);
  cvt_f32_bf16<<<cvtWgtGrid, cvtBlk, 0, stream>>>(Wv, Wvb, nWgt);
  cvt_f32_bf16<<<cvtWgtGrid, cvtBlk, 0, stream>>>(Wo, Wob, nWgt);

  proj_gemm<2><<<gemmGrid, gemmBlk, 0, stream>>>(qb, Wqb, bq, nullptr, Qh,
                                                 M_, D_, D_);
  proj_gemm<2><<<gemmGrid, gemmBlk, 0, stream>>>(kb, Wkb, bk, nullptr, Kh,
                                                 M_, D_, D_);
  proj_gemm<3><<<gemmGrid, gemmBlk, 0, stream>>>(vb, Wvb, bv, nullptr, VhT,
                                                 M_, D_, D_);

  flash_attn<<<faGrid, faBlk, 0, stream>>>(Qh, Kh, VhT, Ab);

  proj_gemm<0><<<gemmGrid, gemmBlk, 0, stream>>>(Ab, Wob, bo, out, nullptr,
                                                 M_, D_, D_);
}
